// DFLFCOS_10909216932636
// MI455X (gfx1250) — matrix operand order repaired and checked by reading, not compiled, not run on hardware
//
#include <hip/hip_runtime.h>
#include <hip/hip_bf16.h>

typedef _Float16 half_t;
typedef __attribute__((ext_vector_type(16))) _Float16 v16h;
typedef __attribute__((ext_vector_type(8)))  _Float16 v8h;
typedef __attribute__((ext_vector_type(8)))  float    v8f;

#define CIN 256
#define TOTALHW 20267
// LDS tile: 3 rows x (32 + 2 halo) cols, 256 ch f16 -> 52,224 bytes
#define LDS_COLS  34
#define LDS_PIX   (3 * LDS_COLS)
#define LDS_HALFS (LDS_PIX * CIN)

// ---------------------------------------------------------------------------
// Pack f32 OIHW weights into WMMA A-fragment layout (f16):
//   packed[((tile*9 + tap)*8 + kb)*32*16 + lane*16 + e]
// A 16x32 k order used here (contiguous k): lane<16: M=lane, K = e
//                        lane>=16: M=lane-16, K = e+16
// ---------------------------------------------------------------------------
__global__ void pack_w_kernel(const float* __restrict__ w, half_t* __restrict__ dst,
                              int tiles, int Ovalid) {
    int t = blockIdx.x * blockDim.x + threadIdx.x;
    int total = tiles * 9 * 8 * 32 * 16;
    if (t >= total) return;
    int e    = t & 15;
    int lane = (t >> 4) & 31;
    int kb   = (t >> 9) & 7;
    int tap  = (t >> 12) % 9;
    int tile = (t >> 12) / 9;
    int hi   = lane >> 4;
    int M    = lane & 15;
    int koff = e + (hi ? 16 : 0);
    int o = tile * 16 + M;
    int c = kb * 32 + koff;
    int ky = tap / 3, kx = tap % 3;
    float v = 0.f;
    if (o < Ovalid) v = w[(((size_t)o * CIN + c) * 3 + ky) * 3 + kx];
    dst[t] = (half_t)v;
}

// f32 NCHW -> f16 NHWC (per-level, B=4, C=256)
__global__ void nchw_to_nhwc_f16(const float* __restrict__ in, half_t* __restrict__ out,
                                 int HW) {
    size_t t = (size_t)blockIdx.x * blockDim.x + threadIdx.x;
    size_t total = (size_t)4 * HW * CIN;
    if (t >= total) return;
    int c = (int)(t & 255);
    size_t rest = t >> 8;
    int p = (int)(rest % HW);
    int b = (int)(rest / HW);
    out[t] = (half_t)in[((size_t)b * CIN + c) * HW + p];
}

// ---------------------------------------------------------------------------
// LDS staging: 32-pixel row-run + 1-pixel halo (3 rows x 34 cols),
// zero-padded out-of-bounds -> hot loop is branch-free.
// ---------------------------------------------------------------------------
__device__ __forceinline__ void stage_tile_lds(half_t* sh, const half_t* __restrict__ inImg,
                                               int y, int x0, int H, int W) {
    // 102 pixels * 256 ch = 26112 halfs = 3264 chunks of 8 halfs (16B)
    for (int i = threadIdx.x; i < LDS_PIX * (CIN / 8); i += 128) {
        int pix  = i >> 5;           // 32 chunks of 8 halfs per pixel
        int cOff = (i & 31) << 3;
        int row  = pix / LDS_COLS;
        int col  = pix - row * LDS_COLS;
        int gy = y - 1 + row;
        int gx = x0 - 1 + col;
        v8h val = {};
        if (gy >= 0 && gy < H && gx >= 0 && gx < W)
            val = *(const v8h*)(inImg + ((size_t)gy * W + gx) * CIN + cOff);
        *(v8h*)(sh + (size_t)pix * CIN + cOff) = val;
    }
}

// 16 out-ch x 32 px: each A fragment feeds TWO WMMAs (A reuse across N tiles).
// 9 taps x 8 K-blocks -> 72 A loads, 144 WMMAs per wave.
__device__ __forceinline__ void conv_accum2(const half_t* sh, const half_t* __restrict__ wpTile,
                                            int lane, v8f& acc0, v8f& acc1) {
    int n  = lane & 15;
    int hi = lane >> 4;
    acc0 = (v8f){};
    acc1 = (v8f){};
    for (int tap = 0; tap < 9; ++tap) {
        int lrow = tap / 3;          // == dy + 1
        int lcol = tap - lrow * 3;   // == dx + 1 offset
        const half_t* b0 = sh + ((size_t)(lrow * LDS_COLS + n + lcol)) * CIN + hi * 16;
        const half_t* b1 = b0 + (size_t)16 * CIN;   // second 16-px N tile
        const half_t* wt = wpTile + ((size_t)tap * 8 * 32 + lane) * 16;
        if (tap < 8)  // prefetch next tap's weight stream (global_prefetch_b8)
            __builtin_prefetch(wpTile + ((size_t)(tap + 1) * 8 * 32 + lane) * 16, 0, 1);
#pragma unroll
        for (int kb = 0; kb < 8; ++kb) {
            v16h afrag  = *(const v16h*)(wt + (size_t)kb * 32 * 16);
            v16h bfrag0 = *(const v16h*)(b0 + kb * 32);
            v16h bfrag1 = *(const v16h*)(b1 + kb * 32);
            acc0 = __builtin_amdgcn_wmma_f32_16x16x32_f16(false, afrag, false, bfrag0,
                                                          (short)0, acc0, false, false);
            acc1 = __builtin_amdgcn_wmma_f32_16x16x32_f16(false, afrag, false, bfrag1,
                                                          (short)0, acc1, false, false);
        }
    }
}

// Tower conv: bias + ReLU, f16 NHWC output (C_out = 256, 16 tiles; 4 per block)
__global__ void __launch_bounds__(128)
conv3x3_wmma_tower(const half_t* __restrict__ in, const half_t* __restrict__ wp,
                   const float* __restrict__ bias, half_t* __restrict__ out,
                   int H, int W) {
    __shared__ half_t sh[LDS_HALFS];
    int lane  = threadIdx.x & 31;
    int wave  = threadIdx.x >> 5;
    int tileS = blockIdx.x;                       // one 32-px spatial tile per block
    int tileC = blockIdx.y * 4 + wave;            // 4 channel tiles per block
    int b     = blockIdx.z;
    int tilesPerRow = (W + 31) >> 5;
    int y  = tileS / tilesPerRow;
    int x0 = (tileS % tilesPerRow) << 5;
    int n  = lane & 15;
    int hi = lane >> 4;

    const half_t* inImg = in + (size_t)b * H * W * CIN;
    stage_tile_lds(sh, inImg, y, x0, H, W);
    __syncthreads();

    v8f acc0, acc1;
    conv_accum2(sh, wp + (size_t)tileC * 9 * 8 * 32 * 16, lane, acc0, acc1);

    size_t rowBase = ((size_t)b * H * W + (size_t)y * W) * CIN;
#pragma unroll
    for (int t = 0; t < 2; ++t) {
        int px = x0 + n + 16 * t;
        if (px < W) {
            const v8f& a = t ? acc1 : acc0;
            size_t row = rowBase + (size_t)px * CIN;
#pragma unroll
            for (int r = 0; r < 8; ++r) {
                int och = tileC * 16 + hi * 8 + r;
                float v = a[r] + bias[och];
                out[row + och] = (half_t)(v > 0.f ? v : 0.f);
            }
        }
    }
}

// Head conv: bias, no ReLU, f32 output rows with arbitrary stride.
// row = b*rowsPerImage + rowOffset + (y*W + px);  out[row*rowStride + och]
__global__ void __launch_bounds__(128)
conv3x3_wmma_head(const half_t* __restrict__ in, const half_t* __restrict__ wp,
                  const float* __restrict__ bias, float* __restrict__ out,
                  int H, int W, int nTilesC,
                  int Ovalid, int rowsPerImage, int rowOffset, int rowStride) {
    __shared__ half_t sh[LDS_HALFS];
    int lane  = threadIdx.x & 31;
    int wave  = threadIdx.x >> 5;
    int tileS = blockIdx.x;
    int tileC = blockIdx.y * 4 + wave;
    int b     = blockIdx.z;
    int tilesPerRow = (W + 31) >> 5;
    int y  = tileS / tilesPerRow;
    int x0 = (tileS % tilesPerRow) << 5;
    int n  = lane & 15;
    int hi = lane >> 4;

    const half_t* inImg = in + (size_t)b * H * W * CIN;
    stage_tile_lds(sh, inImg, y, x0, H, W);
    __syncthreads();

    if (tileC >= nTilesC) return;

    v8f acc0, acc1;
    conv_accum2(sh, wp + (size_t)tileC * 9 * 8 * 32 * 16, lane, acc0, acc1);

#pragma unroll
    for (int t = 0; t < 2; ++t) {
        int px = x0 + n + 16 * t;
        if (px < W) {
            const v8f& a = t ? acc1 : acc0;
            size_t row = (size_t)b * rowsPerImage + rowOffset + (size_t)y * W + px;
#pragma unroll
            for (int r = 0; r < 8; ++r) {
                int och = tileC * 16 + hi * 8 + r;
                if (och < Ovalid)
                    out[row * rowStride + och] = a[r] + bias[och];
            }
        }
    }
}

// DFL decode: softmax over 9 bins, expected value -> out[...][80+s]
__global__ void dfl_kernel(const float* __restrict__ reg, float* __restrict__ out,
                           int HW, int lvlOff) {
    int t = blockIdx.x * blockDim.x + threadIdx.x;
    int total = 4 * HW * 4;                      // B * HW * 4 sides
    if (t >= total) return;
    int s = t & 3;
    int p = (t >> 2) % HW;
    int b = (t >> 2) / HW;
    const float* r = reg + ((size_t)b * HW + p) * 48 + s * 9;
    float m = r[0];
#pragma unroll
    for (int j = 1; j < 9; ++j) m = fmaxf(m, r[j]);
    float den = 0.f, num = 0.f;
#pragma unroll
    for (int j = 0; j < 9; ++j) {
        float e = __expf(r[j] - m);
        den += e;
        num += e * (float)j;
    }
    out[((size_t)b * TOTALHW + lvlOff + p) * 84 + 80 + s] = num / den;
}

extern "C" void kernel_launch(void* const* d_in, const int* in_sizes, int n_in,
                              void* d_out, int out_size, void* d_ws, size_t ws_size,
                              hipStream_t stream) {
    (void)in_sizes; (void)n_in; (void)out_size; (void)ws_size;
    const float* feat[5];
    for (int i = 0; i < 5; ++i) feat[i] = (const float*)d_in[i];
    const float* cls_tw = (const float*)d_in[5];
    const float* cls_tb = (const float*)d_in[6];
    const float* box_tw = (const float*)d_in[7];
    const float* box_tb = (const float*)d_in[8];
    const float* cls_w  = (const float*)d_in[9];
    const float* cls_b  = (const float*)d_in[10];
    const float* box_w  = (const float*)d_in[11];
    const float* box_b  = (const float*)d_in[12];
    float* outp = (float*)d_out;

    // ---- workspace layout -------------------------------------------------
    char* ws = (char*)d_ws;
    size_t off = 0;
    auto alloc = [&](size_t bytes) {
        void* p = ws + off;
        off += (bytes + 255) & ~(size_t)255;
        return p;
    };
    const size_t towerPackB = (size_t)16 * 9 * 8 * 32 * 16 * sizeof(half_t);
    half_t* pwCls[4]; half_t* pwBox[4];
    for (int i = 0; i < 4; ++i) pwCls[i] = (half_t*)alloc(towerPackB);
    for (int i = 0; i < 4; ++i) pwBox[i] = (half_t*)alloc(towerPackB);
    half_t* pwClsHead = (half_t*)alloc((size_t)5 * 9 * 8 * 32 * 16 * sizeof(half_t));
    half_t* pwBoxHead = (half_t*)alloc((size_t)3 * 9 * 8 * 32 * 16 * sizeof(half_t));
    const size_t featB = (size_t)4 * 15200 * CIN * sizeof(half_t);  // max level
    half_t* bufF = (half_t*)alloc(featB);
    half_t* bufA = (half_t*)alloc(featB);
    half_t* bufB = (half_t*)alloc(featB);
    float*  boxBuf = (float*)alloc((size_t)4 * 15200 * 48 * sizeof(float));

    // ---- weight packing ---------------------------------------------------
    {
        int totT = 16 * 9 * 8 * 32 * 16;
        int blk = (totT + 255) / 256;
        for (int i = 0; i < 4; ++i) {
            pack_w_kernel<<<blk, 256, 0, stream>>>(cls_tw + (size_t)i * CIN * CIN * 9, pwCls[i], 16, 256);
            pack_w_kernel<<<blk, 256, 0, stream>>>(box_tw + (size_t)i * CIN * CIN * 9, pwBox[i], 16, 256);
        }
        int totC = 5 * 9 * 8 * 32 * 16;
        pack_w_kernel<<<(totC + 255) / 256, 256, 0, stream>>>(cls_w, pwClsHead, 5, 80);
        int totBx = 3 * 9 * 8 * 32 * 16;
        pack_w_kernel<<<(totBx + 255) / 256, 256, 0, stream>>>(box_w, pwBoxHead, 3, 36);
    }

    static const int LH[5] = {100, 50, 25, 13, 7};
    static const int LW[5] = {152, 76, 38, 19, 10};
    int lvlOff = 0;
    for (int l = 0; l < 5; ++l) {
        int H = LH[l], W = LW[l], HW = H * W;
        int tilesPerRow = (W + 31) / 32;
        int nST = tilesPerRow * H;
        dim3 blk(128);
        dim3 grdTower(nST, 4, 4);                 // 4 ch-tiles per block x grid.y=4 -> 16

        size_t tot = (size_t)4 * HW * CIN;
        nchw_to_nhwc_f16<<<(unsigned)((tot + 255) / 256), 256, 0, stream>>>(feat[l], bufF, HW);

        // cls tower (4 layers) + cls head straight into d_out
        conv3x3_wmma_tower<<<grdTower, blk, 0, stream>>>(bufF, pwCls[0], cls_tb + 0,   bufA, H, W);
        conv3x3_wmma_tower<<<grdTower, blk, 0, stream>>>(bufA, pwCls[1], cls_tb + 256, bufB, H, W);
        conv3x3_wmma_tower<<<grdTower, blk, 0, stream>>>(bufB, pwCls[2], cls_tb + 512, bufA, H, W);
        conv3x3_wmma_tower<<<grdTower, blk, 0, stream>>>(bufA, pwCls[3], cls_tb + 768, bufB, H, W);
        dim3 grdClsHead(nST, 2, 4);               // 8 wave-slots, 5 valid tiles
        conv3x3_wmma_head<<<grdClsHead, blk, 0, stream>>>(bufB, pwClsHead, cls_b, outp,
                                                          H, W, 5, 80, TOTALHW, lvlOff, 84);

        // box tower (4 layers) + box head into 48-ch scratch
        conv3x3_wmma_tower<<<grdTower, blk, 0, stream>>>(bufF, pwBox[0], box_tb + 0,   bufA, H, W);
        conv3x3_wmma_tower<<<grdTower, blk, 0, stream>>>(bufA, pwBox[1], box_tb + 256, bufB, H, W);
        conv3x3_wmma_tower<<<grdTower, blk, 0, stream>>>(bufB, pwBox[2], box_tb + 512, bufA, H, W);
        conv3x3_wmma_tower<<<grdTower, blk, 0, stream>>>(bufA, pwBox[3], box_tb + 768, bufB, H, W);
        dim3 grdBoxHead(nST, 1, 4);               // 4 wave-slots, 3 valid tiles
        conv3x3_wmma_head<<<grdBoxHead, blk, 0, stream>>>(bufB, pwBoxHead, box_b, boxBuf,
                                                          H, W, 3, 36, HW, 0, 48);

        int totD = 4 * HW * 4;
        dfl_kernel<<<(totD + 255) / 256, 256, 0, stream>>>(boxBuf, outp, HW, lvlOff);

        lvlOff += HW;
    }
}
